// JointSupervisedGroundedCoreferencer_45337674776936
// MI455X (gfx1250) — matrix operand order repaired and checked by reading, not compiled, not run on hardware
//
#include <hip/hip_runtime.h>
#include <hip/hip_bf16.h>

typedef __attribute__((ext_vector_type(16))) __bf16 v16bf;
typedef __attribute__((ext_vector_type(8)))  __bf16 v8bf;
typedef __attribute__((ext_vector_type(8)))  float  v8f;

#define DD     1024
#define HH     1024
#define K1     3072
#define BSZ    8
#define NN     40
#define LL     20
#define NPAIRS 780
#define GROWS  (320*160)   // 51200
#define TROWS  (8*780)     // 6240
#define MT     32          // rows per fused block (2 sub-tiles of 16)

static __device__ __forceinline__ unsigned short f2bf(float f){
  unsigned int u = __float_as_uint(f);
  u += 0x7FFFu + ((u >> 16) & 1u);          // round-to-nearest-even
  return (unsigned short)(u >> 16);
}
static __device__ __forceinline__ float bf2f(unsigned short h){
  return __uint_as_float(((unsigned int)h) << 16);
}

// ---------------------------------------------------------------- casts / transposes / pairs
__global__ void cast_bf16_kernel(const float* __restrict__ src,
                                 unsigned short* __restrict__ dst, int n){
  for (int i = blockIdx.x*blockDim.x + threadIdx.x; i < n; i += gridDim.x*blockDim.x)
    dst[i] = f2bf(src[i]);
}

// src: rows x cols (row-major f32)  ->  dst: cols x rows (row-major bf16)
__global__ void transpose_bf16_kernel(const float* __restrict__ src,
                                      unsigned short* __restrict__ dst,
                                      int rows, int cols){
  int total = rows*cols;
  for (int i = blockIdx.x*blockDim.x + threadIdx.x; i < total; i += gridDim.x*blockDim.x){
    int k = i / cols, h = i - k*cols;
    dst[h*rows + k] = f2bf(src[i]);
  }
}

__global__ void triu_pairs_kernel(int* __restrict__ fi, int* __restrict__ si){
  int p = blockIdx.x*blockDim.x + threadIdx.x;
  if (p >= NPAIRS) return;
  int cum = 0;
  for (int f = 0; f < NN-1; ++f){
    int cnt = NN-1-f;
    if (p < cum + cnt){ fi[p] = f; si[p] = f+1+(p-cum); return; }
    cum += cnt;
  }
}

// ---------------------------------------------------------------- small dense GEMM (precompute)
// out[row0+0..15][0..1023] = rows(bf16) @ W(part)  [+ bias]
// WT points at the transposed weight slab: element (n,k) at WT[n*ldw + k], k in [0,1024)
__global__ __launch_bounds__(256)
void gemm_rows_kernel(const unsigned short* __restrict__ rows,  // bf16 [R][1024]
                      const unsigned short* __restrict__ WT,    // bf16 [1024][ldw] (pre-offset)
                      int ldw,
                      const float* __restrict__ bias,           // may be null
                      float* __restrict__ out)                  // f32 [R][1024]
{
  __shared__ unsigned short atile[16*1024];   // 32 KB
  const int tid  = threadIdx.x;
  const int lane = tid & 31;
  const int wave = tid >> 5;
  const int hi   = lane >> 4;
  const int lm   = lane & 15;
  const int c0   = wave * 128;
  const int row0 = blockIdx.x * 16;

  for (int e = tid; e < 16*1024; e += 256)
    atile[e] = rows[(size_t)(row0 + (e >> 10))*DD + (e & 1023)];
  __syncthreads();

  v8f acc[8];
  #pragma unroll
  for (int cb = 0; cb < 8; ++cb){ v8f z = {0.f,0.f,0.f,0.f,0.f,0.f,0.f,0.f}; acc[cb] = z; }

  for (int ks = 0; ks < DD/32; ++ks){
    int koff = ks * 32;
    v8bf alo = *(const v8bf*)&atile[lm*1024 + koff + hi*16];
    v8bf ahi = *(const v8bf*)&atile[lm*1024 + koff + hi*16 + 8];
    v16bf A;
    #pragma unroll
    for (int e = 0; e < 8; ++e){ A[e] = alo[e]; A[8+e] = ahi[e]; }
    #pragma unroll
    for (int cb = 0; cb < 8; ++cb){
      int n = c0 + cb*16 + lm;
      v16bf Bf = *(const v16bf*)&WT[(size_t)n*ldw + koff + hi*16];
      acc[cb] = __builtin_amdgcn_wmma_f32_16x16x32_bf16(false, A, false, Bf,
                                                        (short)0, acc[cb], false, false);
    }
  }
  #pragma unroll
  for (int cb = 0; cb < 8; ++cb){
    int n = c0 + cb*16 + lm;
    float bv = bias ? bias[n] : 0.f;
    #pragma unroll
    for (int r = 0; r < 8; ++r)
      out[(size_t)(row0 + r + 8*hi)*HH + n] = acc[cb][r] + bv;
  }
}

// ---------------------------------------------------------------- fused pair scorer
// For 32 pair-rows: h1 = relu(prod@Wp + addA[ra] + addB[rb]); h2 = relu(h1@W2+b2); sc = h2@W3+b3
__global__ __launch_bounds__(256)
void fused_pair_score_kernel(const unsigned short* __restrict__ srcA,  // bf16 [*,1024]
                             const unsigned short* __restrict__ srcB,
                             const float* __restrict__ addA,           // f32 [*,1024] (f@Wf)
                             const float* __restrict__ addB,           // f32 [*,1024] (v@Ws + b1)
                             const unsigned short* __restrict__ WpT,   // bf16 [1024][ldp] (pre-offset)
                             int ldp,
                             const unsigned short* __restrict__ W2T,   // bf16 [1024][1024]
                             const float* __restrict__ bias2,
                             const float* __restrict__ W3,
                             const float* __restrict__ bias3,
                             const int* __restrict__ fiArr,
                             const int* __restrict__ siArr,
                             int mode,                                 // 0=grounding 1=text
                             float* __restrict__ out)
{
  __shared__ unsigned short ptile[MT*1024];   // 64 KB: product rows -> h1 -> reduction scratch
  const int tid  = threadIdx.x;
  const int lane = tid & 31;
  const int wave = tid >> 5;
  const int hi   = lane >> 4;
  const int lm   = lane & 15;
  const int c0   = wave * 128;
  const int tile = blockIdx.x;

  // ---- stage product rows (bf16) into LDS
  for (int e = tid; e < MT*1024; e += 256){
    int m = e >> 10;
    int k = e & 1023;
    int g = tile*MT + m;
    int ra, rb;
    if (mode == 0){ ra = g / 160; rb = g - ra*160; }
    else { int b = g / NPAIRS; int p = g - b*NPAIRS; ra = b*NN + fiArr[p]; rb = b*NN + siArr[p]; }
    float av = bf2f(srcA[(size_t)ra*DD + k]);
    float bv = bf2f(srcB[(size_t)rb*DD + k]);
    ptile[e] = f2bf(av * bv);
  }
  __syncthreads();

  v8f acc[2][8];
  #pragma unroll
  for (int mt = 0; mt < 2; ++mt)
    #pragma unroll
    for (int cb = 0; cb < 8; ++cb){ v8f z = {0.f,0.f,0.f,0.f,0.f,0.f,0.f,0.f}; acc[mt][cb] = z; }

  // ---- phase 1: prod(32x1024) @ Wp — each B fragment reused for 2 row sub-tiles
  for (int ks = 0; ks < DD/32; ++ks){
    int koff = ks * 32;
    v16bf A0, A1;
    {
      v8bf lo = *(const v8bf*)&ptile[lm*1024 + koff + hi*16];
      v8bf hi8 = *(const v8bf*)&ptile[lm*1024 + koff + hi*16 + 8];
      #pragma unroll
      for (int e = 0; e < 8; ++e){ A0[e] = lo[e]; A0[8+e] = hi8[e]; }
    }
    {
      v8bf lo = *(const v8bf*)&ptile[(16+lm)*1024 + koff + hi*16];
      v8bf hi8 = *(const v8bf*)&ptile[(16+lm)*1024 + koff + hi*16 + 8];
      #pragma unroll
      for (int e = 0; e < 8; ++e){ A1[e] = lo[e]; A1[8+e] = hi8[e]; }
    }
    #pragma unroll
    for (int cb = 0; cb < 8; ++cb){
      int n = c0 + cb*16 + lm;
      v16bf Bf = *(const v16bf*)&WpT[(size_t)n*ldp + koff + hi*16];
      acc[0][cb] = __builtin_amdgcn_wmma_f32_16x16x32_bf16(false, A0, false, Bf,
                                                           (short)0, acc[0][cb], false, false);
      acc[1][cb] = __builtin_amdgcn_wmma_f32_16x16x32_bf16(false, A1, false, Bf,
                                                           (short)0, acc[1][cb], false, false);
    }
  }

  // ---- epilogue 1: + addA[ra] + addB[rb], relu
  int ra8[2][8], rb8[2][8];
  #pragma unroll
  for (int mt = 0; mt < 2; ++mt)
    #pragma unroll
    for (int r = 0; r < 8; ++r){
      int g = tile*MT + mt*16 + r + 8*hi;
      if (mode == 0){ ra8[mt][r] = g / 160; rb8[mt][r] = g - (g/160)*160; }
      else { int b = g / NPAIRS; int p = g - b*NPAIRS;
             ra8[mt][r] = b*NN + fiArr[p]; rb8[mt][r] = b*NN + siArr[p]; }
    }
  #pragma unroll
  for (int mt = 0; mt < 2; ++mt)
    #pragma unroll
    for (int cb = 0; cb < 8; ++cb){
      int n = c0 + cb*16 + lm;
      #pragma unroll
      for (int r = 0; r < 8; ++r){
        float v = acc[mt][cb][r] + addA[(size_t)ra8[mt][r]*HH + n]
                                 + addB[(size_t)rb8[mt][r]*HH + n];
        acc[mt][cb][r] = v > 0.f ? v : 0.f;
      }
    }
  __syncthreads();                 // all waves done reading ptile (products)

  // ---- store h1 (bf16) over ptile
  #pragma unroll
  for (int mt = 0; mt < 2; ++mt)
    #pragma unroll
    for (int cb = 0; cb < 8; ++cb){
      int n = c0 + cb*16 + lm;
      #pragma unroll
      for (int r = 0; r < 8; ++r)
        ptile[(mt*16 + r + 8*hi)*1024 + n] = f2bf(acc[mt][cb][r]);
    }
  __syncthreads();

  // ---- phase 2: h1(32x1024) @ W2
  #pragma unroll
  for (int mt = 0; mt < 2; ++mt)
    #pragma unroll
    for (int cb = 0; cb < 8; ++cb){ v8f z = {0.f,0.f,0.f,0.f,0.f,0.f,0.f,0.f}; acc[mt][cb] = z; }
  for (int ks = 0; ks < HH/32; ++ks){
    int koff = ks * 32;
    v16bf A0, A1;
    {
      v8bf lo = *(const v8bf*)&ptile[lm*1024 + koff + hi*16];
      v8bf hi8 = *(const v8bf*)&ptile[lm*1024 + koff + hi*16 + 8];
      #pragma unroll
      for (int e = 0; e < 8; ++e){ A0[e] = lo[e]; A0[8+e] = hi8[e]; }
    }
    {
      v8bf lo = *(const v8bf*)&ptile[(16+lm)*1024 + koff + hi*16];
      v8bf hi8 = *(const v8bf*)&ptile[(16+lm)*1024 + koff + hi*16 + 8];
      #pragma unroll
      for (int e = 0; e < 8; ++e){ A1[e] = lo[e]; A1[8+e] = hi8[e]; }
    }
    #pragma unroll
    for (int cb = 0; cb < 8; ++cb){
      int n = c0 + cb*16 + lm;
      v16bf Bf = *(const v16bf*)&W2T[(size_t)n*HH + koff + hi*16];
      acc[0][cb] = __builtin_amdgcn_wmma_f32_16x16x32_bf16(false, A0, false, Bf,
                                                           (short)0, acc[0][cb], false, false);
      acc[1][cb] = __builtin_amdgcn_wmma_f32_16x16x32_bf16(false, A1, false, Bf,
                                                           (short)0, acc[1][cb], false, false);
    }
  }
  __syncthreads();                 // all waves done reading h1

  // ---- phase 3: bias2 + relu, dot with W3, reduce
  float* red = (float*)ptile;      // 32 floats, overlays dead h1
  if (tid < MT) red[tid] = 0.f;
  __syncthreads();

  float part[2][8];
  #pragma unroll
  for (int mt = 0; mt < 2; ++mt)
    #pragma unroll
    for (int r = 0; r < 8; ++r) part[mt][r] = 0.f;
  #pragma unroll
  for (int mt = 0; mt < 2; ++mt)
    #pragma unroll
    for (int cb = 0; cb < 8; ++cb){
      int n = c0 + cb*16 + lm;
      float bv = bias2[n];
      float w3 = W3[n];
      #pragma unroll
      for (int r = 0; r < 8; ++r){
        float v = acc[mt][cb][r] + bv;
        v = v > 0.f ? v : 0.f;
        part[mt][r] += v * w3;
      }
    }
  #pragma unroll
  for (int mt = 0; mt < 2; ++mt)
    #pragma unroll
    for (int r = 0; r < 8; ++r)
      atomicAdd(&red[mt*16 + r + 8*hi], part[mt][r]);
  __syncthreads();

  if (tid < MT){
    int g = tile*MT + tid;
    float sc = red[tid] + bias3[0];
    if (mode == 0){
      int i = g / 160, j = g - (g/160)*160;
      int b1i = i / NN, n = i - b1i*NN;
      int b2i = j / LL, l = j - b2i*LL;
      out[((b1i*BSZ + b2i)*NN + n)*LL + l] = sc;   // (B,B,N,L) flat
    } else {
      out[g] = sc;                                  // (B,780) flat
    }
  }
}

// ---------------------------------------------------------------- loss
__global__ void loss_kernel(const float* __restrict__ gsc, float* __restrict__ out0){
  __shared__ float S[64];
  int t = threadIdx.x;
  if (t < 64){
    float s = 0.f;
    const float* p = gsc + t*800;                  // (b1,b2) block of 40*20 contiguous
    for (int k = 0; k < 800; ++k) s += p[k];
    S[t] = s;
  }
  __syncthreads();
  if (t == 0){
    float loss = 0.f;
    for (int r = 0; r < 8; ++r){                   // log_softmax over rows
      float mx = -1e30f;
      for (int c = 0; c < 8; ++c) mx = fmaxf(mx, S[r*8+c]);
      float se = 0.f;
      for (int c = 0; c < 8; ++c) se += __expf(S[r*8+c] - mx);
      float lse = mx + __logf(se);
      for (int c = 0; c < 8; ++c) loss -= (S[r*8+c] - lse);
    }
    for (int c = 0; c < 8; ++c){                   // log_softmax over columns (S.T rows)
      float mx = -1e30f;
      for (int r = 0; r < 8; ++r) mx = fmaxf(mx, S[r*8+c]);
      float se = 0.f;
      for (int r = 0; r < 8; ++r) se += __expf(S[r*8+c] - mx);
      float lse = mx + __logf(se);
      for (int r = 0; r < 8; ++r) loss -= (S[r*8+c] - lse);
    }
    out0[0] = loss / (float)BSZ;
  }
}

// ---------------------------------------------------------------- launch
extern "C" void kernel_launch(void* const* d_in, const int* in_sizes, int n_in,
                              void* d_out, int out_size, void* d_ws, size_t ws_size,
                              hipStream_t stream){
  const float* span = (const float*)d_in[0];
  const float* img  = (const float*)d_in[1];
  const float* tW1  = (const float*)d_in[4];
  const float* tb1  = (const float*)d_in[5];
  const float* tW2  = (const float*)d_in[6];
  const float* tb2  = (const float*)d_in[7];
  const float* tW3  = (const float*)d_in[8];
  const float* tb3  = (const float*)d_in[9];
  const float* gW1  = (const float*)d_in[10];
  const float* gb1  = (const float*)d_in[11];
  const float* gW2  = (const float*)d_in[12];
  const float* gb2  = (const float*)d_in[13];
  const float* gW3  = (const float*)d_in[14];
  const float* gb3  = (const float*)d_in[15];
  float* out = (float*)d_out;

  char* ws = (char*)d_ws;
  size_t off = 0;
  auto alloc = [&](size_t bytes)->void*{
    void* p = ws + off;
    off += (bytes + 255) & ~(size_t)255;
    return p;
  };
  unsigned short* f_bf = (unsigned short*)alloc((size_t)320*1024*2);
  unsigned short* v_bf = (unsigned short*)alloc((size_t)160*1024*2);
  unsigned short* gW1T = (unsigned short*)alloc((size_t)1024*3072*2);
  unsigned short* tW1T = (unsigned short*)alloc((size_t)1024*3072*2);
  unsigned short* gW2T = (unsigned short*)alloc((size_t)1024*1024*2);
  unsigned short* tW2T = (unsigned short*)alloc((size_t)1024*1024*2);
  int*   fiArr = (int*)alloc((size_t)NPAIRS*4);
  int*   siArr = (int*)alloc((size_t)NPAIRS*4);
  float* FW = (float*)alloc((size_t)320*1024*4);   // f @ gWf
  float* VW = (float*)alloc((size_t)160*1024*4);   // v @ gWs + gb1
  float* TF = (float*)alloc((size_t)320*1024*4);   // span @ tWf
  float* TS = (float*)alloc((size_t)320*1024*4);   // span @ tWs + tb1

  cast_bf16_kernel<<<512, 256, 0, stream>>>(span, f_bf, 320*1024);
  cast_bf16_kernel<<<256, 256, 0, stream>>>(img,  v_bf, 160*1024);
  transpose_bf16_kernel<<<2048, 256, 0, stream>>>(gW1, gW1T, 3072, 1024);
  transpose_bf16_kernel<<<2048, 256, 0, stream>>>(tW1, tW1T, 3072, 1024);
  transpose_bf16_kernel<<<1024, 256, 0, stream>>>(gW2, gW2T, 1024, 1024);
  transpose_bf16_kernel<<<1024, 256, 0, stream>>>(tW2, tW2T, 1024, 1024);
  triu_pairs_kernel<<<4, 256, 0, stream>>>(fiArr, siArr);

  // single-index precomputations (K=3072 -> 1024 in the pair phase)
  gemm_rows_kernel<<<20, 256, 0, stream>>>(f_bf, gW1T,        K1, nullptr, FW);
  gemm_rows_kernel<<<10, 256, 0, stream>>>(v_bf, gW1T + 1024, K1, gb1,     VW);
  gemm_rows_kernel<<<20, 256, 0, stream>>>(f_bf, tW1T,        K1, nullptr, TF);
  gemm_rows_kernel<<<20, 256, 0, stream>>>(f_bf, tW1T + 1024, K1, tb1,     TS);

  // grounding: 51200 rows / 32 = 1600 tiles; output at d_out[1..51200]
  fused_pair_score_kernel<<<GROWS/MT, 256, 0, stream>>>(
      f_bf, v_bf, FW, VW, gW1T + 2048, K1, gW2T, gb2, gW3, gb3,
      fiArr, siArr, 0, out + 1);
  // text: 6240 rows / 32 = 195 tiles; output at d_out[1+51200 ..]
  fused_pair_score_kernel<<<TROWS/MT, 256, 0, stream>>>(
      f_bf, f_bf, TF, TS, tW1T + 2048, K1, tW2T, tb2, tW3, tb3,
      fiArr, siArr, 1, out + 1 + GROWS);

  loss_kernel<<<1, 64, 0, stream>>>(out + 1, out);
}
